// PrototypeClassifier_34411277975719
// MI455X (gfx1250) — matrix operand order repaired and checked by reading, not compiled, not run on hardware
//
#include <hip/hip_runtime.h>

// ---------------------------------------------------------------------------
// PrototypeClassifier: d[n,c] = ||x_n||^2 + ||p_c||^2 - 2 * (x @ p^T)[n,c]
// N=16384, C=4096, D=1024, fp32 in/out.
//
// 3-stage pipeline per K-step (BK=32):
//   ASYNC DMA  : global_load_async_to_lds_b128 stages raw f32 tile kt+2
//   CONVERT    : f32 -> packed bf16 hi/lo tiles (each element once), gated by
//                counted s_wait_asynccnt (own-lane data only, no barrier)
//   COMPUTE    : ds_load_b128 fragment gathers + 24x v_wmma_f32_16x16x32_bf16
//                (bf16 hi/lo 3-product fp32 emulation, f32 accumulate)
// Single __syncthreads per iteration publishes the bf16 tiles.
// ---------------------------------------------------------------------------

typedef __attribute__((ext_vector_type(16))) __bf16 v16bf;
typedef __attribute__((ext_vector_type(8)))  float  v8f;
typedef __attribute__((ext_vector_type(4)))  float  f32x4;
typedef __attribute__((ext_vector_type(4)))  __bf16 bf16x4;
typedef __attribute__((ext_vector_type(4)))  unsigned int u32x4;

#define DIM_N 16384
#define DIM_C 4096
#define DIM_D 1024

#define BM 128
#define BN 128
#define BK 32
#define LDK 40   // padded bf16 row pitch: 80B -> conflict-free b128 gathers
#define KT (DIM_D / BK)

// dynamic LDS: f32 staging (2x16KB x2) + bf16 hi/lo tiles (4 x 2 x 10KB) = 144KB
#define SMEM_BYTES ((2 * BM * BK + 2 * BN * BK) * 4 + 4 * (2 * BM * LDK) * 2)

union FragU { v16bf v; u32x4 q[2]; };

__device__ __forceinline__ v8f wmma_bf16(v16bf a, v16bf b, v8f c) {
  // (neg_a, A, neg_b, B, c_mod, C, reuse_a, reuse_b)
  return __builtin_amdgcn_wmma_f32_16x16x32_bf16(false, a, false, b,
                                                 (short)0, c, false, false);
}

// Low 32 bits of a generic pointer to LDS = LDS byte offset
// (hardware adds LDS_BASE to the async instruction's VDST address).
__device__ __forceinline__ unsigned lds_off(const void* p) {
  return (unsigned)(unsigned long long)p;
}

// ---------------------------------------------------------------------------
// Row squared-norms: one wave (32 lanes) per row, coalesced, shfl reduction.
// ---------------------------------------------------------------------------
__global__ void __launch_bounds__(256)
rownorm_kernel(const float* __restrict__ m, float* __restrict__ out,
               int rows, int cols) {
  const int lane = threadIdx.x & 31;
  const int wid  = threadIdx.x >> 5;
  const int row  = blockIdx.x * 8 + wid;
  if (row >= rows) return;
  const float* r = m + (size_t)row * cols;
  float s = 0.0f;
  for (int k = lane; k < cols; k += 32) {
    float v = r[k];
    s += v * v;
  }
#pragma unroll
  for (int off = 16; off > 0; off >>= 1)
    s += __shfl_xor(s, off, 32);
  if (lane == 0) out[row] = s;
}

// ---------------------------------------------------------------------------
// Fused distance GEMM.
// Block: 256 threads = 8 waves, 4(M) x 2(N); block tile 128x128;
// wave tile 32x64 = 2x4 accumulators of 16x16 f32.
// ---------------------------------------------------------------------------
__global__ void __launch_bounds__(256)
proto_dist_kernel(const float* __restrict__ X, const float* __restrict__ P,
                  const float* __restrict__ xsq, const float* __restrict__ psq,
                  float* __restrict__ out) {
  extern __shared__ __align__(16) char smem[];
  float*  As = (float*)smem;              // [2][BM*BK]  f32 staging (ping-pong)
  float*  Bs = As + 2 * BM * BK;          // [2][BN*BK]
  __bf16* Ah = (__bf16*)(Bs + 2 * BN * BK);   // [2][BM*LDK] packed hi
  __bf16* Al = Ah + 2 * BM * LDK;             // [2][BM*LDK] packed lo
  __bf16* Bh = Al + 2 * BM * LDK;
  __bf16* Bl = Bh + 2 * BN * LDK;

  const int tid  = threadIdx.x;
  const int lane = tid & 31;
  const int wid  = tid >> 5;
  const int wm   = wid >> 1;   // 0..3  (M direction)
  const int wn   = wid & 1;    // 0..1  (N direction)

  const int n0 = blockIdx.y * BM;
  const int c0 = blockIdx.x * BN;

  // DMA one 128x32 f32 tile of X and P into staging buffer `stg` (8 async ops).
  auto async_stage = [&](int k0, int stg) {
    const float* sa = As + stg * BM * BK;
    const float* sb = Bs + stg * BN * BK;
#pragma unroll
    for (int t = 0; t < 4; ++t) {
      int i  = tid + t * 256;      // float4 index in 128x32 tile
      int r  = i >> 3;             // row 0..127
      int c4 = i & 7;              // float4 column 0..7
      unsigned la = lds_off(sa + i * 4);
      unsigned lb = lds_off(sb + i * 4);
      const float* ga = X + (size_t)(n0 + r) * DIM_D + k0 + c4 * 4;
      const float* gb = P + (size_t)(c0 + r) * DIM_D + k0 + c4 * 4;
      asm volatile("global_load_async_to_lds_b128 %0, %1, off"
                   :: "v"(la), "v"(ga) : "memory");
      asm volatile("global_load_async_to_lds_b128 %0, %1, off"
                   :: "v"(lb), "v"(gb) : "memory");
    }
  };

  // Each thread converts exactly the bytes it DMA'd (own async ops only).
  auto convert_tile = [&](int buf) {
    const float* sa  = As + buf * BM * BK;
    const float* sb  = Bs + buf * BN * BK;
    __bf16* pah = Ah + buf * BM * LDK;
    __bf16* pal = Al + buf * BM * LDK;
    __bf16* pbh = Bh + buf * BN * LDK;
    __bf16* pbl = Bl + buf * BN * LDK;
#pragma unroll
    for (int t = 0; t < 4; ++t) {
      int i  = tid + t * 256;
      int r  = i >> 3;
      int c4 = i & 7;
      f32x4 fa = *(const f32x4*)(sa + i * 4);
      f32x4 fb = *(const f32x4*)(sb + i * 4);
      bf16x4 ha, la, hb, lb;
#pragma unroll
      for (int j = 0; j < 4; ++j) {
        __bf16 h = (__bf16)fa[j];
        ha[j] = h;
        la[j] = (__bf16)(fa[j] - (float)h);   // lo via fma_mix
        __bf16 g = (__bf16)fb[j];
        hb[j] = g;
        lb[j] = (__bf16)(fb[j] - (float)g);
      }
      *(bf16x4*)(pah + r * LDK + c4 * 4) = ha;   // ds_store_b64, 8B aligned
      *(bf16x4*)(pal + r * LDK + c4 * 4) = la;
      *(bf16x4*)(pbh + r * LDK + c4 * 4) = hb;
      *(bf16x4*)(pbl + r * LDK + c4 * 4) = lb;
    }
  };

  // 16-bit A 16x32 layout: lane m holds K=0..7 & 16..23; lane m+16: 8..15 & 24..31
  auto load_fragA = [&](const __bf16* base, int mt) -> v16bf {
    int row = wm * 32 + mt * 16 + (lane & 15);
    int k   = (lane < 16) ? 0 : 8;
    FragU f;
    f.q[0] = *(const u32x4*)(base + row * LDK + k);
    f.q[1] = *(const u32x4*)(base + row * LDK + k + 16);
    return f.v;
  };

  // 16-bit B 32x16 layout: lane n holds K=0..7 & 16..23; lane n+16: 8..15 & 24..31
  auto load_fragB = [&](const __bf16* base, int nt) -> v16bf {
    int row = wn * 64 + nt * 16 + (lane & 15);
    int k   = (lane < 16) ? 0 : 8;
    FragU f;
    f.q[0] = *(const u32x4*)(base + row * LDK + k);
    f.q[1] = *(const u32x4*)(base + row * LDK + k + 16);
    return f.v;
  };

  v8f acc[2][4];
#pragma unroll
  for (int mt = 0; mt < 2; ++mt)
#pragma unroll
    for (int nt = 0; nt < 4; ++nt)
      acc[mt][nt] = (v8f){0.f, 0.f, 0.f, 0.f, 0.f, 0.f, 0.f, 0.f};

  // Prologue: DMA tile0, convert it (own data -> just drain own asynccnt),
  // start DMA of tile1, publish bf16 buf0.
  async_stage(0, 0);
  asm volatile("s_wait_asynccnt 0x0" ::: "memory");
  convert_tile(0);
  async_stage(BK, 1);
  __syncthreads();

  for (int kt = 0; kt < KT; ++kt) {
    const int cur = kt & 1;
    if (kt + 2 < KT) async_stage((kt + 2) * BK, cur);  // staging[cur] is free

    // ---- compute tile kt from bf16 buffers [cur] ----
    const __bf16* a_h = Ah + cur * BM * LDK;
    const __bf16* a_l = Al + cur * BM * LDK;
    const __bf16* b_h = Bh + cur * BN * LDK;
    const __bf16* b_l = Bl + cur * BN * LDK;
    v16bf ah0 = load_fragA(a_h, 0), al0 = load_fragA(a_l, 0);
    v16bf ah1 = load_fragA(a_h, 1), al1 = load_fragA(a_l, 1);
#pragma unroll
    for (int nt = 0; nt < 4; ++nt) {
      v16bf bh = load_fragB(b_h, nt);
      v16bf bl = load_fragB(b_l, nt);
      acc[0][nt] = wmma_bf16(ah0, bh, acc[0][nt]);
      acc[0][nt] = wmma_bf16(ah0, bl, acc[0][nt]);
      acc[0][nt] = wmma_bf16(al0, bh, acc[0][nt]);
      acc[1][nt] = wmma_bf16(ah1, bh, acc[1][nt]);
      acc[1][nt] = wmma_bf16(ah1, bl, acc[1][nt]);
      acc[1][nt] = wmma_bf16(al1, bh, acc[1][nt]);
    }

    // ---- convert tile kt+1 (overlaps with WMMA drain) ----
    if (kt + 1 < KT) {
      if (kt + 2 < KT) {
        // two batches in flight; async loads complete in order ->
        // <=8 outstanding means batch kt+1 has fully landed
        asm volatile("s_wait_asynccnt 0x8" ::: "memory");
      } else {
        asm volatile("s_wait_asynccnt 0x0" ::: "memory");
      }
      convert_tile((kt + 1) & 1);
    }

    __syncthreads();   // publish bf16[kt+1], protect bf16[cur] until read
  }

  // Fused epilogue: d = ||x||^2 + ||p||^2 - 2*cross
  // C/D layout: lanes 0-15 -> M=r, N=lane; lanes 16-31 -> M=r+8, N=lane-16
#pragma unroll
  for (int mt = 0; mt < 2; ++mt) {
#pragma unroll
    for (int nt = 0; nt < 4; ++nt) {
      int c = c0 + wn * 64 + nt * 16 + (lane & 15);
      float pq = psq[c];
#pragma unroll
      for (int r = 0; r < 8; ++r) {
        int n = n0 + wm * 32 + mt * 16 + ((lane < 16) ? r : r + 8);
        out[(size_t)n * DIM_C + c] = xsq[n] + pq - 2.0f * acc[mt][nt][r];
      }
    }
  }
}

// ---------------------------------------------------------------------------
extern "C" void kernel_launch(void* const* d_in, const int* in_sizes, int n_in,
                              void* d_out, int out_size, void* d_ws, size_t ws_size,
                              hipStream_t stream) {
  (void)in_sizes; (void)n_in; (void)out_size; (void)ws_size;
  const float* x = (const float*)d_in[0];   // [N, D] f32
  const float* p = (const float*)d_in[1];   // [C, D] f32
  float* out = (float*)d_out;               // [N, C] f32

  float* xsq = (float*)d_ws;                // N floats
  float* psq = xsq + DIM_N;                 // C floats (80 KB of ws total)

  rownorm_kernel<<<DIM_N / 8, 256, 0, stream>>>(x, xsq, DIM_N, DIM_D);
  rownorm_kernel<<<DIM_C / 8, 256, 0, stream>>>(p, psq, DIM_C, DIM_D);

  dim3 grid(DIM_C / BN, DIM_N / BM);        // (32, 128)
  proto_dist_kernel<<<grid, 256, SMEM_BYTES, stream>>>(x, p, xsq, psq, out);
}
